// multihead_self_attention_3616362463610
// MI455X (gfx1250) — matrix operand order repaired and checked by reading, not compiled, not run on hardware
//
#include <hip/hip_runtime.h>

#define D_MODEL 1024
#define NUM_HEADS 16
#define DKH 64            // head dim
#define NB 4              // batch
#define NS 2048           // seq len
#define BS (NB * NS)      // 8192 rows

typedef __attribute__((ext_vector_type(16))) _Float16     v16h;
typedef __attribute__((ext_vector_type(8)))  float        v8f;
typedef __attribute__((ext_vector_type(4)))  unsigned int u32x4;
typedef __attribute__((ext_vector_type(8)))  int          i32x8;
typedef __attribute__((ext_vector_type(4)))  int          i32x4;
typedef __attribute__((ext_vector_type(4)))  _Float16     h4;
typedef __attribute__((ext_vector_type(4)))  float        f4;

#if defined(__has_builtin)
#if __has_builtin(__builtin_amdgcn_tensor_load_to_lds) && __has_builtin(__builtin_amdgcn_s_wait_tensorcnt)
#define HAVE_TDM 1
#endif
#endif
#ifndef HAVE_TDM
#define HAVE_TDM 0
#endif

union FragU { v16h v; u32x4 u[2]; };

__device__ __forceinline__ v8f v8f_zero() {
    v8f z;
#pragma unroll
    for (int i = 0; i < 8; ++i) z[i] = 0.0f;
    return z;
}

__device__ __forceinline__ v8f wmma16(v16h a, v16h b, v8f c) {
    // D = A(16x32 f16) * B(32x16 f16) + C(16x16 f32)
    return __builtin_amdgcn_wmma_f32_16x16x32_f16(
        /*neg_a=*/false, a, /*neg_b=*/false, b,
        /*c_mod=*/(short)0, c, /*reuse_a=*/false, /*reuse_b=*/false);
}

// A-fragment (16x32 f16, row-major, lda in elements): two b128 loads per lane.
__device__ __forceinline__ v16h fragA(const _Float16* __restrict__ base, int lda,
                                      int row0, int k0, int lane) {
    const int m  = lane & 15;
    const int kb = (lane >> 4) << 4;
    const _Float16* p = base + (size_t)(row0 + m) * lda + k0;
    FragU f;
    f.u[0] = *reinterpret_cast<const u32x4*>(p + kb);
    f.u[1] = *reinterpret_cast<const u32x4*>(p + kb + 8);
    return f.v;
}

// B-fragment (32x16 f16). Storage row n = column n of B (16 contiguous K per half).
__device__ __forceinline__ v16h fragB(const _Float16* __restrict__ base, int ldb,
                                      int n0, int k0, int lane) {
    const int n  = lane & 15;
    const int kb = (lane >> 4) << 4;
    const _Float16* p = base + (size_t)(n0 + n) * ldb + k0 + kb;
    FragU f;
    f.u[0] = *reinterpret_cast<const u32x4*>(p);
    f.u[1] = *reinterpret_cast<const u32x4*>(p + 8);
    return f.v;
}

#if HAVE_TDM
// TDM: DMA a 64-row x 32-col f16 tile (row stride = D_MODEL elements) into LDS.
// D# group layout per CDNA5 ISA (2D tensor, data_size=2B).
// This toolchain exposes the 6-arg builtin:
//   (uint32x4 g0, int32x8 g1, int32x4 g2, int32x4 g3, int32x8 pad, i32 cpol)
__device__ __forceinline__ void tdm_load_tile(_Float16* ldsdst, const _Float16* gsrc) {
    unsigned long long ga = (unsigned long long)(uintptr_t)gsrc;
    u32x4 g0;
    g0[0] = 1u;                                   // count=1, user descriptor
    g0[1] = (unsigned)(uintptr_t)ldsdst;          // lds_addr (low 32 bits of generic ptr)
    g0[2] = (unsigned)ga;                         // global_addr[31:0]
    g0[3] = (unsigned)(ga >> 32) | 0x80000000u;   // global_addr[56:32] | type=2 (bits 127:126)
    i32x8 g1;
    g1[0] = (int)(1u << 16);                      // workgroup_mask=0, data_size=1 (2 bytes)
    g1[1] = (int)((D_MODEL & 0xffffu) << 16);     // tensor_dim0[15:0] (=1024)
    g1[2] = (int)((D_MODEL >> 16) | (64u << 16)); // tensor_dim0[31:16] | tensor_dim1[15:0]
    g1[3] = (int)(32u << 16);                     // tensor_dim1[31:16]=0 | tile_dim0=32
    g1[4] = (int)64u;                             // tile_dim1=64, tile_dim2=0
    g1[5] = (int)D_MODEL;                         // tensor_dim0_stride[31:0]
    g1[6] = 0;                                    // stride hi | dim1_stride lo (unused, 2D)
    g1[7] = 0;
    i32x4 z4 = {0, 0, 0, 0};
    i32x8 z8 = {0, 0, 0, 0, 0, 0, 0, 0};
    __builtin_amdgcn_tensor_load_to_lds(g0, g1, z4, z4, z8, 0);
}
#endif

// fallback / cooperative stage of a 64x32 f16 tile by the whole 256-thread block
__device__ __forceinline__ void coop_stage_tile(_Float16* ldsdst, const _Float16* gsrc) {
    int t = threadIdx.x;
    int row = t >> 2, q = (t & 3) * 8;            // 256 threads x 8 halves = 64x32
    *reinterpret_cast<u32x4*>(ldsdst + row * 32 + q) =
        *reinterpret_cast<const u32x4*>(gsrc + (size_t)row * D_MODEL + q);
}

// ---------------------------------------------------------------- convert f32 -> f16
__global__ void __launch_bounds__(256) cvt_f16_kernel(const float* __restrict__ src,
                                                      _Float16* __restrict__ dst, int n4) {
    int i = blockIdx.x * blockDim.x + threadIdx.x;
    if (i < n4) {
        f4 v = reinterpret_cast<const f4*>(src)[i];
        h4 o;
#pragma unroll
        for (int j = 0; j < 4; ++j) o[j] = (_Float16)v[j];
        reinterpret_cast<h4*>(dst)[i] = o;
    }
}

// ---------------------------------------------------------------- fused QKV GEMM + RoPE
// grid: (BS/256, NUM_HEADS, 3), block 256: 8 waves, each 32 rows x 64 cols.
// Weight tile (64 x 32 per K-chunk) staged once per block via TDM, double-buffered.
__global__ void __launch_bounds__(256) qkv_rope_kernel(
    const _Float16* __restrict__ xh,
    const _Float16* __restrict__ wqh,
    const _Float16* __restrict__ wkh,
    const _Float16* __restrict__ wvh,
    const int* __restrict__ tokpos,
    _Float16* __restrict__ qb,      // [B,H,S,DK]
    _Float16* __restrict__ kb_out,  // [B,H,S,DK]
    _Float16* __restrict__ vT)      // [B,H,DK,S]
{
    __shared__ alignas(16) _Float16 wlds[2][64][32];   // 8 KB double buffer

    const int lane = threadIdx.x & 31;
    const int wave = threadIdx.x >> 5;
    const int row0 = blockIdx.x * 256 + wave * 32;
    const int h    = blockIdx.y;
    const int widx = blockIdx.z;
    const int nlo = lane & 15, lh = lane >> 4;
    const int kb2 = lh << 4;

    const _Float16* w = (widx == 0) ? wqh : (widx == 1) ? wkh : wvh;
    const _Float16* wbase = w + (size_t)h * DKH * D_MODEL;

    v8f acc[2][4];
#pragma unroll
    for (int rt = 0; rt < 2; ++rt)
#pragma unroll
        for (int dt = 0; dt < 4; ++dt) acc[rt][dt] = v8f_zero();

    // prologue: stage chunk 0 into buffer 0
#if HAVE_TDM
    if (wave == 0) tdm_load_tile(&wlds[0][0][0], wbase);
#else
    coop_stage_tile(&wlds[0][0][0], wbase);
#endif

    for (int k0 = 0; k0 < D_MODEL; k0 += 32) {
        const int cur = (k0 >> 5) & 1;
#if HAVE_TDM
        if (wave == 0) {
            if (k0 + 32 < D_MODEL) {
                tdm_load_tile(&wlds[cur ^ 1][0][0], wbase + k0 + 32);
                __builtin_amdgcn_s_wait_tensorcnt(1);   // current tile complete
            } else {
                __builtin_amdgcn_s_wait_tensorcnt(0);
            }
        }
#else
        if (k0 + 32 < D_MODEL) coop_stage_tile(&wlds[cur ^ 1][0][0], wbase + k0 + 32);
#endif
        __syncthreads();   // tile `cur` visible to all waves

        __builtin_prefetch((const void*)(xh + (size_t)row0 * D_MODEL + k0 + 64), 0, 1);
        v16h a0 = fragA(xh, D_MODEL, row0,      k0, lane);
        v16h a1 = fragA(xh, D_MODEL, row0 + 16, k0, lane);
#pragma unroll
        for (int dt = 0; dt < 4; ++dt) {
            const _Float16* bp = &wlds[cur][dt * 16 + nlo][kb2];
            FragU f;
            f.u[0] = *reinterpret_cast<const u32x4*>(bp);
            f.u[1] = *reinterpret_cast<const u32x4*>(bp + 8);
            acc[0][dt] = wmma16(a0, f.v, acc[0][dt]);
            acc[1][dt] = wmma16(a1, f.v, acc[1][dt]);
        }
        __syncthreads();   // all reads of `cur` done before it is overwritten
    }

#pragma unroll
    for (int rt = 0; rt < 2; ++rt) {
        const int rowb = row0 + rt * 16;
        int brow[8], srow[8];
        float posr[8];
#pragma unroll
        for (int r = 0; r < 8; ++r) {
            int row = rowb + r + 8 * lh;
            brow[r] = row >> 11;
            srow[r] = row & (NS - 1);
            posr[r] = (float)tokpos[srow[r]];
        }
        if (widx == 2) {
            // V transposed: [B,H,DK,S] so PV B-fragments are contiguous key runs.
#pragma unroll
            for (int dt = 0; dt < 4; ++dt) {
                int dk = dt * 16 + nlo;
#pragma unroll
                for (int r = 0; r < 8; ++r) {
                    size_t idx = (((size_t)brow[r] * NUM_HEADS + h) * DKH + dk) * NS + srow[r];
                    vT[idx] = (_Float16)acc[rt][dt][r];
                }
            }
        } else {
            _Float16* dst = (widx == 0) ? qb : kb_out;
#pragma unroll
            for (int dt = 0; dt < 4; ++dt) {
                int   dk       = dt * 16 + nlo;
                float inv_freq = __expf(-(9.21034037f / 64.0f) * (float)(dk & ~1));
                float sgn      = (dk & 1) ? 1.0f : -1.0f;
#pragma unroll
                for (int r = 0; r < 8; ++r) {
                    float ang = posr[r] * inv_freq;
                    float sn, cs;
                    __sincosf(ang, &sn, &cs);
                    float v       = acc[rt][dt][r];
                    float partner = __shfl_xor(v, 1, 32);  // even<->odd rope pair
                    float outv    = v * cs + sgn * partner * sn;
                    size_t idx = (((size_t)brow[r] * NUM_HEADS + h) * NS + srow[r]) * DKH + dk;
                    dst[idx] = (_Float16)outv;
                }
            }
        }
    }
}

// ---------------------------------------------------------------- causal flash attention
// grid: (NS/16/8, B*H), block 256: each wave owns one 16-row query tile.
__global__ void __launch_bounds__(256) attn_kernel(
    const _Float16* __restrict__ qb,
    const _Float16* __restrict__ kbuf,
    const _Float16* __restrict__ vT,
    _Float16* __restrict__ ob)      // [B,S,D_MODEL]
{
    __shared__ alignas(16) _Float16 plds[8][16][32];   // per-wave P relayout buffer

    const int lane = threadIdx.x & 31;
    const int wave = threadIdx.x >> 5;
    const int nlo = lane & 15, lh = lane >> 4;
    const int q0 = (blockIdx.x * 8 + wave) * 16;
    const int bh = blockIdx.y;
    const int b = bh >> 4, h = bh & 15;

    const _Float16* qh = qb   + (size_t)bh * NS * DKH;
    const _Float16* kh = kbuf + (size_t)bh * NS * DKH;
    const _Float16* vh = vT   + (size_t)bh * DKH * NS;

    // Q fragments: loaded once, reused for every key tile.
    v16h aq0 = fragA(qh, DKH, q0, 0, lane);
    v16h aq1 = fragA(qh, DKH, q0, 32, lane);

    v8f oacc[4];
#pragma unroll
    for (int dt = 0; dt < 4; ++dt) oacc[dt] = v8f_zero();
    float mr[8], lr[8];
#pragma unroll
    for (int r = 0; r < 8; ++r) { mr[r] = -3.0e38f; lr[r] = 0.0f; }

    for (int kt = 0; kt < q0 + 16; kt += 32) {   // causal: keys <= q0+15
        v8f s0 = v8f_zero(), s1 = v8f_zero();
        {
            v16h b0 = fragB(kh, DKH, kt, 0, lane);
            v16h b1 = fragB(kh, DKH, kt, 32, lane);
            s0 = wmma16(aq0, b0, s0);
            s0 = wmma16(aq1, b1, s0);
            v16h b2 = fragB(kh, DKH, kt + 16, 0, lane);
            v16h b3 = fragB(kh, DKH, kt + 16, 32, lane);
            s1 = wmma16(aq0, b2, s1);
            s1 = wmma16(aq1, b3, s1);
        }
#pragma unroll
        for (int r = 0; r < 8; ++r) {
            int qrow = q0 + r + 8 * lh;
            s0[r] = (kt + nlo      <= qrow) ? s0[r] * 0.125f : -3.0e38f;
            s1[r] = (kt + 16 + nlo <= qrow) ? s1[r] * 0.125f : -3.0e38f;
        }
        // online softmax (per-row stats via 16-lane shfl reductions)
#pragma unroll
        for (int r = 0; r < 8; ++r) {
            float mx = fmaxf(s0[r], s1[r]);
#pragma unroll
            for (int off = 1; off < 16; off <<= 1) mx = fmaxf(mx, __shfl_xor(mx, off, 32));
            float mnew = fmaxf(mr[r], mx);
            float fr = __expf(mr[r] - mnew);
            float p0 = __expf(s0[r] - mnew);
            float p1 = __expf(s1[r] - mnew);
            s0[r] = p0; s1[r] = p1;
            float rs = p0 + p1;
#pragma unroll
            for (int off = 1; off < 16; off <<= 1) rs += __shfl_xor(rs, off, 32);
            lr[r] = lr[r] * fr + rs;
            mr[r] = mnew;
#pragma unroll
            for (int dt = 0; dt < 4; ++dt) oacc[dt][r] *= fr;
        }
        // P: C/D layout -> A layout through per-wave LDS (same-wave DS ops are in-order)
#pragma unroll
        for (int r = 0; r < 8; ++r) {
            plds[wave][r + 8 * lh][nlo]      = (_Float16)s0[r];
            plds[wave][r + 8 * lh][16 + nlo] = (_Float16)s1[r];
        }
        __asm__ volatile("s_wait_dscnt 0" ::: "memory");
        FragU fp;
        {
            const int kb = lh << 4;
            fp.u[0] = *reinterpret_cast<const u32x4*>(&plds[wave][nlo][kb]);
            fp.u[1] = *reinterpret_cast<const u32x4*>(&plds[wave][nlo][kb + 8]);
        }
#pragma unroll
        for (int dt = 0; dt < 4; ++dt) {
            v16h bv = fragB(vh, NS, dt * 16, kt, lane);
            oacc[dt] = wmma16(fp.v, bv, oacc[dt]);
        }
        __asm__ volatile("s_wait_dscnt 0" ::: "memory");
    }

#pragma unroll
    for (int r = 0; r < 8; ++r) {
        float inv = 1.0f / lr[r];
        int s = q0 + r + 8 * lh;
        size_t base = ((size_t)b * NS + s) * D_MODEL + h * DKH;
#pragma unroll
        for (int dt = 0; dt < 4; ++dt)
            ob[base + dt * 16 + nlo] = (_Float16)(oacc[dt][r] * inv);
    }
}

// ---------------------------------------------------------------- output projection
// grid: (BS/256, D_MODEL/64), block 256: 8 waves, each 32 rows x 64 cols.
__global__ void __launch_bounds__(256) oproj_kernel(
    const _Float16* __restrict__ ob,
    const _Float16* __restrict__ woh,
    float* __restrict__ out)
{
    __shared__ alignas(16) _Float16 wlds[2][64][32];

    const int lane = threadIdx.x & 31;
    const int wave = threadIdx.x >> 5;
    const int row0 = blockIdx.x * 256 + wave * 32;
    const int e0   = blockIdx.y * 64;
    const int nlo = lane & 15, lh = lane >> 4;
    const int kb2 = lh << 4;

    const _Float16* wbase = woh + (size_t)e0 * D_MODEL;

    v8f acc[2][4];
#pragma unroll
    for (int rt = 0; rt < 2; ++rt)
#pragma unroll
        for (int dt = 0; dt < 4; ++dt) acc[rt][dt] = v8f_zero();

#if HAVE_TDM
    if (wave == 0) tdm_load_tile(&wlds[0][0][0], wbase);
#else
    coop_stage_tile(&wlds[0][0][0], wbase);
#endif

    for (int k0 = 0; k0 < D_MODEL; k0 += 32) {
        const int cur = (k0 >> 5) & 1;
#if HAVE_TDM
        if (wave == 0) {
            if (k0 + 32 < D_MODEL) {
                tdm_load_tile(&wlds[cur ^ 1][0][0], wbase + k0 + 32);
                __builtin_amdgcn_s_wait_tensorcnt(1);
            } else {
                __builtin_amdgcn_s_wait_tensorcnt(0);
            }
        }
#else
        if (k0 + 32 < D_MODEL) coop_stage_tile(&wlds[cur ^ 1][0][0], wbase + k0 + 32);
#endif
        __syncthreads();

        __builtin_prefetch((const void*)(ob + (size_t)row0 * D_MODEL + k0 + 64), 0, 1);
        v16h a0 = fragA(ob, D_MODEL, row0,      k0, lane);
        v16h a1 = fragA(ob, D_MODEL, row0 + 16, k0, lane);
#pragma unroll
        for (int dt = 0; dt < 4; ++dt) {
            const _Float16* bp = &wlds[cur][dt * 16 + nlo][kb2];
            FragU f;
            f.u[0] = *reinterpret_cast<const u32x4*>(bp);
            f.u[1] = *reinterpret_cast<const u32x4*>(bp + 8);
            acc[0][dt] = wmma16(a0, f.v, acc[0][dt]);
            acc[1][dt] = wmma16(a1, f.v, acc[1][dt]);
        }
        __syncthreads();
    }

#pragma unroll
    for (int rt = 0; rt < 2; ++rt) {
#pragma unroll
        for (int dt = 0; dt < 4; ++dt) {
#pragma unroll
            for (int r = 0; r < 8; ++r) {
                size_t idx = (size_t)(row0 + rt * 16 + r + 8 * lh) * D_MODEL + e0 + dt * 16 + nlo;
                out[idx] = acc[rt][dt][r];
            }
        }
    }
}

// ---------------------------------------------------------------- launcher
extern "C" void kernel_launch(void* const* d_in, const int* in_sizes, int n_in,
                              void* d_out, int out_size, void* d_ws, size_t ws_size,
                              hipStream_t stream) {
    (void)in_sizes; (void)n_in; (void)out_size; (void)ws_size;
    const float* x  = (const float*)d_in[0];
    const int*  tok = (const int*)d_in[1];
    const float* wq = (const float*)d_in[2];
    const float* wk = (const float*)d_in[3];
    const float* wv = (const float*)d_in[4];
    const float* wo = (const float*)d_in[5];
    float* out = (float*)d_out;

    char* ws = (char*)d_ws;
    size_t off = 0;
    _Float16* xh  = (_Float16*)(ws + off); off += (size_t)BS * D_MODEL * 2;       // 16 MB
    _Float16* wqh = (_Float16*)(ws + off); off += (size_t)D_MODEL * D_MODEL * 2;  //  2 MB
    _Float16* wkh = (_Float16*)(ws + off); off += (size_t)D_MODEL * D_MODEL * 2;
    _Float16* wvh = (_Float16*)(ws + off); off += (size_t)D_MODEL * D_MODEL * 2;
    _Float16* woh = (_Float16*)(ws + off); off += (size_t)D_MODEL * D_MODEL * 2;
    _Float16* qb  = (_Float16*)(ws + off); off += (size_t)BS * D_MODEL * 2;       // [B,H,S,DK]
    _Float16* kb  = (_Float16*)(ws + off); off += (size_t)BS * D_MODEL * 2;       // [B,H,S,DK]
    _Float16* vT  = (_Float16*)(ws + off); off += (size_t)BS * D_MODEL * 2;       // [B,H,DK,S]
    _Float16* ob  = (_Float16*)(ws + off); off += (size_t)BS * D_MODEL * 2;       // [B,S,D]

    {
        int n4 = BS * D_MODEL / 4;
        cvt_f16_kernel<<<(n4 + 255) / 256, 256, 0, stream>>>(x, xh, n4);
        int w4 = D_MODEL * D_MODEL / 4;
        cvt_f16_kernel<<<(w4 + 255) / 256, 256, 0, stream>>>(wq, wqh, w4);
        cvt_f16_kernel<<<(w4 + 255) / 256, 256, 0, stream>>>(wk, wkh, w4);
        cvt_f16_kernel<<<(w4 + 255) / 256, 256, 0, stream>>>(wv, wvh, w4);
        cvt_f16_kernel<<<(w4 + 255) / 256, 256, 0, stream>>>(wo, woh, w4);
    }
    qkv_rope_kernel<<<dim3(BS / 256, NUM_HEADS, 3), 256, 0, stream>>>(
        xh, wqh, wkh, wvh, tok, qb, kb, vT);
    attn_kernel<<<dim3(NS / 16 / 8, NB * NUM_HEADS), 256, 0, stream>>>(qb, kb, vT, ob);
    oproj_kernel<<<dim3(BS / 256, D_MODEL / 64), 256, 0, stream>>>(ob, woh, out);
}
